// QueryKeySelfAttention_20332375179636
// MI455X (gfx1250) — compile-verified, matrix operand order checked by reading, not run on hardware
//
#include <hip/hip_runtime.h>
#include <cstdint>
#include <cstddef>

// ---------------------------------------------------------------------------
// Types for CDNA5 WMMA (wave32): 16x16x32 bf16 -> f32
// ---------------------------------------------------------------------------
typedef __attribute__((ext_vector_type(16))) __bf16          v16bf;
typedef __attribute__((ext_vector_type(16))) unsigned short  v16u;
typedef __attribute__((ext_vector_type(8)))  float           v8f;
typedef __attribute__((ext_vector_type(4)))  unsigned int    u32x4;
typedef __attribute__((ext_vector_type(8)))  unsigned int    u32x8;

#define B_   16
#define S_   1024
#define D_   1024
#define H_   16
#define HD_  64

static __device__ __forceinline__ unsigned short f32_to_bf16(float f) {
  unsigned u = __float_as_uint(f);
  unsigned r = 0x7FFFu + ((u >> 16) & 1u);   // round-to-nearest-even
  return (unsigned short)((u + r) >> 16);
}

// Packed f32x2 -> bf16x2 (RNE). D[15:0]=cvt(a), D[31:16]=cvt(b).
static __device__ __forceinline__ unsigned cvt_pk_bf16(float a, float b) {
  unsigned d;
  asm("v_cvt_pk_bf16_f32 %0, %1, %2" : "=v"(d) : "v"(a), "v"(b));
  return d;
}

static __device__ __forceinline__ v16bf ld_frag(const unsigned short* p) {
  return __builtin_bit_cast(v16bf, *(const v16u*)p);
}

static __device__ __forceinline__ v8f wmma_bf16(v16bf a, v16bf b, v8f c) {
  return __builtin_amdgcn_wmma_f32_16x16x32_bf16(
      false, a, false, b, (short)0, c, false, false);
}

// ---------------------------------------------------------------------------
// TDM: 2D tensor_load_to_lds. D# groups per ISA 8.3/8.4 (VADDR2/3 = null).
// Units: data_size=1 -> 2-byte elements; dims/strides in elements.
// ---------------------------------------------------------------------------
static __device__ __forceinline__ void tdm_load_2d(
    const void* gaddr, unsigned lds_off,
    unsigned tensor_d0, unsigned tensor_d1, unsigned long long stride0,
    unsigned tile_d0, unsigned tile_d1) {
  const unsigned long long ga = (unsigned long long)(uintptr_t)gaddr;
  u32x4 g0;
  g0[0] = 1u;                                          // count=1, user D#
  g0[1] = lds_off;                                     // lds_addr (bytes)
  g0[2] = (unsigned)(ga & 0xFFFFFFFFu);                // global_addr[31:0]
  g0[3] = (unsigned)((ga >> 32) & 0x1FFFFFFu) | (2u << 30);  // [56:32] | type=2
  u32x8 g1;
  g1[0] = (1u << 16);                                  // data_size=1 (2B)
  g1[1] = (tensor_d0 & 0xFFFFu) << 16;                 // tensor_dim0 lo16
  g1[2] = (tensor_d0 >> 16) | ((tensor_d1 & 0xFFFFu) << 16);
  g1[3] = (tensor_d1 >> 16) | (tile_d0 << 16);         // tile_dim0
  g1[4] = tile_d1 & 0xFFFFu;                           // tile_dim1 (tile_dim2=0)
  g1[5] = (unsigned)(stride0 & 0xFFFFFFFFu);           // tensor_dim0_stride lo
  g1[6] = (unsigned)((stride0 >> 32) & 0xFFFFu);       // stride hi16 (dim1 stride=0)
  g1[7] = 0u;
  asm volatile("tensor_load_to_lds %0, %1, null, null"
               :: "s"(g0), "s"(g1)
               : "memory");
}

// ---------------------------------------------------------------------------
// Kernel 1: Vt[b,h,d,s] = bf16(X[b,s,h*64+d])  (LDS 64x64 tile transpose)
// ---------------------------------------------------------------------------
__global__ __launch_bounds__(256) void v_transpose_kernel(
    const float* __restrict__ X, unsigned short* __restrict__ vt) {
  __shared__ __align__(32) unsigned short tile[64][66];  // padded banks
  const int idx = blockIdx.x;
  const int st = idx & 15;
  const int bh = idx >> 4;
  const int b  = bh >> 4;
  const int h  = bh & 15;
  const int s0 = st * 64;

  const int r = threadIdx.x >> 2;
  const int c = (threadIdx.x & 3) * 16;
  const float4* src =
      (const float4*)(X + (size_t)(b * S_ + s0 + r) * D_ + h * HD_ + c);
  unsigned* trow = (unsigned*)&tile[r][c];   // dword-aligned (c even, 66 even)
#pragma unroll
  for (int v = 0; v < 4; ++v) {
    float4 f = src[v];
    trow[v * 2 + 0] = cvt_pk_bf16(f.x, f.y);
    trow[v * 2 + 1] = cvt_pk_bf16(f.z, f.w);
  }
  __syncthreads();

  const int d  = threadIdx.x >> 2;
  const int sc = (threadIdx.x & 3) * 16;
  v16u pack;
#pragma unroll
  for (int i = 0; i < 16; ++i) pack[i] = tile[sc + i][d];
  *(v16u*)(vt + (size_t)(bh * HD_ + d) * S_ + s0 + sc) = pack;
}

// ---------------------------------------------------------------------------
// Kernel 2: qk = X @ W^T + b; Q,K written bf16 in head layout [B,H,S,HD]
// Block tile 128(M) x 128(N), 8 waves (2x4), wave tile 64x32, K-step 32.
// ---------------------------------------------------------------------------
__global__ __launch_bounds__(256) void qk_proj_kernel(
    const float* __restrict__ X, const float* __restrict__ W,
    const float* __restrict__ bias,
    unsigned short* __restrict__ qb, unsigned short* __restrict__ kb) {
  __shared__ __align__(32) unsigned short a_lds[128 * 32];
  __shared__ __align__(32) unsigned short b_lds[128 * 32];

  const int tid  = threadIdx.x;
  const int lane = tid & 31;
  const int wave = tid >> 5;
  const int lr   = lane & 15;
  const int kc   = (lane >> 4) * 16;
  const int hi8  = (lane >> 4) * 8;
  const int wm   = wave & 1;
  const int wn   = wave >> 1;
  const int Mb   = blockIdx.x * 128;
  const int Nb   = blockIdx.y * 128;

  v8f acc[4][2];
#pragma unroll
  for (int m = 0; m < 4; ++m)
#pragma unroll
    for (int n = 0; n < 2; ++n) acc[m][n] = {};

  const int ldr = tid >> 1;           // staged row 0..127
  const int ldc = (tid & 1) * 16;     // staged col half
  unsigned* a_dw = (unsigned*)a_lds;
  unsigned* b_dw = (unsigned*)b_lds;
  const int dwo = ldr * 16 + (ldc >> 1);   // dword offset in 128x32 bf16 tile

  for (int kt = 0; kt < D_; kt += 32) {
    __syncthreads();
    {
      const float4* xa = (const float4*)(X + (size_t)(Mb + ldr) * D_ + kt + ldc);
      const float4* wa = (const float4*)(W + (size_t)(Nb + ldr) * D_ + kt + ldc);
      unsigned dx[8], dw[8];
#pragma unroll
      for (int v = 0; v < 4; ++v) {
        float4 fx = xa[v];
        float4 fw = wa[v];
        dx[v * 2 + 0] = cvt_pk_bf16(fx.x, fx.y);
        dx[v * 2 + 1] = cvt_pk_bf16(fx.z, fx.w);
        dw[v * 2 + 0] = cvt_pk_bf16(fw.x, fw.y);
        dw[v * 2 + 1] = cvt_pk_bf16(fw.z, fw.w);
      }
      *(uint4*)(a_dw + dwo)     = make_uint4(dx[0], dx[1], dx[2], dx[3]);
      *(uint4*)(a_dw + dwo + 4) = make_uint4(dx[4], dx[5], dx[6], dx[7]);
      *(uint4*)(b_dw + dwo)     = make_uint4(dw[0], dw[1], dw[2], dw[3]);
      *(uint4*)(b_dw + dwo + 4) = make_uint4(dw[4], dw[5], dw[6], dw[7]);
    }
    __syncthreads();

    v16bf afr[4], bfr[2];
#pragma unroll
    for (int mt = 0; mt < 4; ++mt)
      afr[mt] = ld_frag(&a_lds[(wm * 64 + mt * 16 + lr) * 32 + kc]);
#pragma unroll
    for (int nt = 0; nt < 2; ++nt)
      bfr[nt] = ld_frag(&b_lds[(wn * 32 + nt * 16 + lr) * 32 + kc]);
#pragma unroll
    for (int mt = 0; mt < 4; ++mt)
#pragma unroll
      for (int nt = 0; nt < 2; ++nt)
        acc[mt][nt] = wmma_bf16(afr[mt], bfr[nt], acc[mt][nt]);
  }

  // Epilogue: + bias, scatter bf16 into Q or K head layout [B,H,S,HD]
#pragma unroll
  for (int nt = 0; nt < 2; ++nt) {
    const int colg = Nb + wn * 32 + nt * 16 + lr;
    const float bv = bias[colg];
    const bool isQ = colg < D_;
    const int  c2  = isQ ? colg : colg - D_;
    const int  h   = c2 >> 6;
    const int  d   = c2 & 63;
    unsigned short* dst = isQ ? qb : kb;
#pragma unroll
    for (int mt = 0; mt < 4; ++mt)
#pragma unroll
      for (int j = 0; j < 8; ++j) {
        const int Mg = Mb + wm * 64 + mt * 16 + j + hi8;
        const int bb = Mg >> 10;
        const int ss = Mg & 1023;
        dst[((size_t)(bb * H_ + h) * S_ + ss) * HD_ + d] =
            f32_to_bf16(acc[mt][nt][j] + bv);
      }
  }
}

// ---------------------------------------------------------------------------
// Kernel 3: flash attention. 4 waves x 32 q-rows; kv tiles of 32; HD=64.
// K/V tiles staged to LDS by TDM (double-buffered), fragments via ds_load.
// ---------------------------------------------------------------------------
__global__ __launch_bounds__(128) void attn_kernel(
    const unsigned short* __restrict__ qb, const unsigned short* __restrict__ kbuf,
    const unsigned short* __restrict__ vt, float* __restrict__ out) {
  __shared__ __align__(32) unsigned short k_lds[2][32 * 64];  // kv x d
  __shared__ __align__(32) unsigned short v_lds[2][64 * 32];  // d  x kv
  __shared__ __align__(32) unsigned short p_lds[4][32 * 32];  // per-wave P

  const int tid  = threadIdx.x;
  const int lane = tid & 31;
  const int wave = tid >> 5;
  const int lr   = lane & 15;
  const int kc   = (lane >> 4) * 16;
  const int hi8  = (lane >> 4) * 8;
  const int bh   = blockIdx.y;
  const int b    = bh >> 4;
  const int h    = bh & 15;
  const int q0   = blockIdx.x * 128 + wave * 32;

  const unsigned short* Q = qb   + (size_t)bh * (S_ * HD_);
  const unsigned short* K = kbuf + (size_t)bh * (S_ * HD_);
  const unsigned short* V = vt   + (size_t)bh * (HD_ * S_);
  unsigned short* pl = p_lds[wave];
  const float scale = 0.125f;   // 1/sqrt(64)

  // Resident Q fragments: 2 m-tiles x 2 K-halves of head dim
  v16bf qf[2][2];
#pragma unroll
  for (int t = 0; t < 2; ++t)
#pragma unroll
    for (int ks = 0; ks < 2; ++ks)
      qf[t][ks] = ld_frag(Q + (size_t)(q0 + t * 16 + lr) * HD_ + ks * 32 + kc);

  v8f   o[2][4];
  float mrun[2][8], lrun[2][8];
#pragma unroll
  for (int t = 0; t < 2; ++t) {
#pragma unroll
    for (int dt = 0; dt < 4; ++dt) o[t][dt] = {};
#pragma unroll
    for (int j = 0; j < 8; ++j) { mrun[t][j] = -1e30f; lrun[t][j] = 0.0f; }
  }

  // Prologue: wave 0 issues TDM for kv=0 into buffer 0
  if (wave == 0) {
    tdm_load_2d(K, (unsigned)(uintptr_t)&k_lds[0][0], HD_, S_, HD_, HD_, 32);
    tdm_load_2d(V, (unsigned)(uintptr_t)&v_lds[0][0], S_, HD_, S_, 32, HD_);
  }

  for (int kv = 0; kv < S_; kv += 32) {
    const int ib = (kv >> 5) & 1;
    if (wave == 0) __builtin_amdgcn_s_wait_tensorcnt(0);
    __syncthreads();   // publish buffer ib; also proves prior reads of ib^1 done
    if (wave == 0 && (kv + 32) < S_) {
      const int nkv = kv + 32;
      tdm_load_2d(K + (size_t)nkv * HD_, (unsigned)(uintptr_t)&k_lds[ib ^ 1][0],
                  HD_, S_, HD_, HD_, 32);
      tdm_load_2d(V + nkv, (unsigned)(uintptr_t)&v_lds[ib ^ 1][0],
                  S_, HD_, S_, 32, HD_);
    }
    const unsigned short* kt = &k_lds[ib][0];
    const unsigned short* vtile = &v_lds[ib][0];

    // ---- S = Q K^T (8 WMMAs), K fragments from LDS ----
    v8f sa[2][2];
#pragma unroll
    for (int t = 0; t < 2; ++t)
#pragma unroll
      for (int nt = 0; nt < 2; ++nt) sa[t][nt] = {};
#pragma unroll
    for (int ks = 0; ks < 2; ++ks)
#pragma unroll
      for (int nt = 0; nt < 2; ++nt) {
        v16bf kf = ld_frag(kt + (nt * 16 + lr) * HD_ + ks * 32 + kc);
#pragma unroll
        for (int t = 0; t < 2; ++t)
          sa[t][nt] = wmma_bf16(qf[t][ks], kf, sa[t][nt]);
      }

    // ---- online softmax ----
#pragma unroll
    for (int t = 0; t < 2; ++t)
#pragma unroll
      for (int j = 0; j < 8; ++j) {
        float s0 = sa[t][0][j] * scale;
        float s1 = sa[t][1][j] * scale;
        float mx = fmaxf(s0, s1);
        mx = fmaxf(mx, __shfl_xor(mx, 1, 32));
        mx = fmaxf(mx, __shfl_xor(mx, 2, 32));
        mx = fmaxf(mx, __shfl_xor(mx, 4, 32));
        mx = fmaxf(mx, __shfl_xor(mx, 8, 32));
        const float mnew = fmaxf(mrun[t][j], mx);
        const float corr = __expf(mrun[t][j] - mnew);
        mrun[t][j] = mnew;
        const float p0 = __expf(s0 - mnew);
        const float p1 = __expf(s1 - mnew);
        float rs = p0 + p1;
        rs += __shfl_xor(rs, 1, 32);
        rs += __shfl_xor(rs, 2, 32);
        rs += __shfl_xor(rs, 4, 32);
        rs += __shfl_xor(rs, 8, 32);
        lrun[t][j] = lrun[t][j] * corr + rs;
        const int r = t * 16 + j + hi8;
        pl[r * 32 + lr]      = f32_to_bf16(p0);
        pl[r * 32 + 16 + lr] = f32_to_bf16(p1);
#pragma unroll
        for (int dt = 0; dt < 4; ++dt) o[t][dt][j] *= corr;
      }

    asm volatile("s_wait_dscnt 0" ::: "memory");  // same-wave LDS RAW fence

    // ---- O += P V (8 WMMAs), V fragments from LDS ----
    v16bf vf[4];
#pragma unroll
    for (int dt = 0; dt < 4; ++dt)
      vf[dt] = ld_frag(vtile + (dt * 16 + lr) * 32 + kc);
#pragma unroll
    for (int t = 0; t < 2; ++t) {
      v16bf pf = ld_frag(pl + (t * 16 + lr) * 32 + kc);
#pragma unroll
      for (int dt = 0; dt < 4; ++dt)
        o[t][dt] = wmma_bf16(pf, vf[dt], o[t][dt]);
    }
  }

  // ---- epilogue: O / l, f32 store to [B,S,D] ----
#pragma unroll
  for (int t = 0; t < 2; ++t)
#pragma unroll
    for (int j = 0; j < 8; ++j) {
      const float inv = 1.0f / lrun[t][j];
      const int srow = q0 + t * 16 + j + hi8;
      float* op = out + (size_t)(b * S_ + srow) * D_ + h * HD_;
#pragma unroll
      for (int dt = 0; dt < 4; ++dt) op[dt * 16 + lr] = o[t][dt][j] * inv;
    }
}

// ---------------------------------------------------------------------------
extern "C" void kernel_launch(void* const* d_in, const int* in_sizes, int n_in,
                              void* d_out, int out_size, void* d_ws, size_t ws_size,
                              hipStream_t stream) {
  (void)in_sizes; (void)n_in; (void)out_size; (void)ws_size;
  const float* X    = (const float*)d_in[0];   // [B,S,D]
  const float* W    = (const float*)d_in[1];   // [2D,D]
  const float* bias = (const float*)d_in[2];   // [2D]
  float* out = (float*)d_out;                  // [B,S,D] f32

  const size_t headElems = (size_t)B_ * H_ * S_ * HD_;
  unsigned short* qb = (unsigned short*)d_ws;
  unsigned short* kb = qb + headElems;
  unsigned short* vt = kb + headElems;

  v_transpose_kernel<<<dim3(B_ * H_ * (S_ / 64)), 256, 0, stream>>>(X, vt);
  qk_proj_kernel<<<dim3((B_ * S_) / 128, (2 * D_) / 128), 256, 0, stream>>>(
      X, W, bias, qb, kb);
  attn_kernel<<<dim3(S_ / 128, B_ * H_), 128, 0, stream>>>(qb, kb, vt, out);
}
